// Gatv2CN_16312285790929
// MI455X (gfx1250) — matrix operand order repaired and checked by reading, not compiled, not run on hardware
//
#include <hip/hip_runtime.h>

typedef __bf16 bf16;
typedef __attribute__((ext_vector_type(16))) __bf16 v16bf;
typedef __attribute__((ext_vector_type(8)))  float  v8f;

#define DM   128   // H*HID
#define HDS  4
#define HIDC 32
#define NEGS 0.2f

// ---------- helpers ----------
__device__ __forceinline__ unsigned f2ord(float f) {
  unsigned b = __float_as_uint(f);
  return (b & 0x80000000u) ? ~b : (b | 0x80000000u);
}
__device__ __forceinline__ float ord2f(unsigned u) {
  unsigned b = (u & 0x80000000u) ? (u & 0x7FFFFFFFu) : ~u;
  return __uint_as_float(b);
}

// ---------- small utility kernels ----------
__global__ void k_cvt_bf16(const float* __restrict__ src, bf16* __restrict__ dst, int total) {
  int i = blockIdx.x * blockDim.x + threadIdx.x;
  if (i < total) dst[i] = (bf16)src[i];
}

// Pre-pack W (K x 128, row-major f32) into WMMA-B fragment order:
// fragment (kb/32, tileN) for lane L holds 16 contiguous bf16:
//   n = tileN*16 + (L&15), k = kb + (L>=16 ? 16 : 0) + elem
__global__ void k_pack_w(const float* __restrict__ W, bf16* __restrict__ out, int K) {
  int i = blockIdx.x * blockDim.x + threadIdx.x;
  if (i >= K * DM) return;
  int elem  = i & 15;
  int lane  = (i >> 4) & 31;
  int tileN = (i >> 9) & 7;
  int kb    = (i >> 12) << 5;
  int n = tileN * 16 + (lane & 15);
  int k = kb + ((lane >> 4) ? 16 : 0) + elem;
  out[i] = (bf16)W[(size_t)k * DM + n];
}

__global__ void k_loop_accum(const int* __restrict__ ei, const float* __restrict__ ea,
                             float* __restrict__ loop_attr, float* __restrict__ cnt, int E) {
  int e = blockIdx.x * blockDim.x + threadIdx.x;
  if (e >= E) return;
  int d = ei[E + e];
  atomicAdd(&cnt[d], 1.0f);
#pragma unroll
  for (int j = 0; j < 4; ++j) atomicAdd(&loop_attr[(size_t)d * 4 + j], ea[(size_t)e * 4 + j]);
}

__global__ void k_loop_final(float* __restrict__ loop_attr, const float* __restrict__ cnt, int N) {
  int i = blockIdx.x * blockDim.x + threadIdx.x;
  if (i >= N * 4) return;
  float c = fmaxf(cnt[i >> 2], 1.0f);
  loop_attr[i] /= c;
}

// ---------- WMMA GEMM: one wave owns a 16x128 output strip ----------
__device__ __forceinline__ v16bf load_a_frag(const bf16* __restrict__ X, int ldx, int nrows,
                                             int rowBase, int kBase, int lane) {
  int m = rowBase + (lane & 15);
  if (m >= nrows) m = nrows - 1;
  int koff = (lane >> 4) ? 16 : 0;
  const bf16* row = X + (size_t)m * ldx + kBase + koff;
  v16bf a;
#pragma unroll
  for (int i = 0; i < 8; ++i) a[i] = row[i];
#pragma unroll
  for (int i = 0; i < 8; ++i) a[8 + i] = row[8 + i];
  return a;
}

__device__ __forceinline__ v16bf load_b_packed(const bf16* __restrict__ Wp, int kb, int tileN,
                                               int lane) {
  // fragment-contiguous: ((kb/32)*8 + tileN)*32 + lane, 16 bf16 each (32B aligned)
  const v16bf* p =
      (const v16bf*)(Wp + (((size_t)(kb >> 5) * 8 + tileN) * 32 + lane) * 16);
  return *p;
}

template <int KV>
__global__ __launch_bounds__(32, 1) __attribute__((amdgpu_waves_per_eu(1)))
void k_gemm_wmma(const bf16* __restrict__ X, const bf16* __restrict__ Wlp,
                 const bf16* __restrict__ Wrp, const float* __restrict__ bl,
                 const float* __restrict__ br, float* __restrict__ xl,
                 float* __restrict__ xr, int nrows) {
  int rowBase = blockIdx.x * 16;
  int lane = threadIdx.x;  // wave32: one wave per block
  v8f cl[8], cr[8];
#pragma unroll
  for (int t = 0; t < 8; ++t) { cl[t] = (v8f){}; cr[t] = (v8f){}; }

#pragma unroll
  for (int kb = 0; kb < KV; kb += 32) {
    v16bf a = load_a_frag(X, KV, nrows, rowBase, kb, lane);
#pragma unroll
    for (int t = 0; t < 8; ++t) {
      v16bf bL = load_b_packed(Wlp, kb, t, lane);
      v16bf bR = load_b_packed(Wrp, kb, t, lane);
      cl[t] = __builtin_amdgcn_wmma_f32_16x16x32_bf16(false, a, false, bL, (short)0, cl[t],
                                                      false, false);
      cr[t] = __builtin_amdgcn_wmma_f32_16x16x32_bf16(false, a, false, bR, (short)0, cr[t],
                                                      false, false);
    }
  }

  int lan = lane & 15;
  int mo = (lane >> 4) ? 8 : 0;
#pragma unroll
  for (int t = 0; t < 8; ++t) {
    int n = t * 16 + lan;
    float bLn = bl[n], bRn = br[n];
#pragma unroll
    for (int j = 0; j < 8; ++j) {
      int m = rowBase + mo + j;
      if (m < nrows) {
        xl[(size_t)m * DM + n] = cl[t][j] + bLn;
        xr[(size_t)m * DM + n] = cr[t][j] + bRn;
      }
    }
  }
}

// ---------- edge kernels: one thread per (edge, head) ----------
__device__ __forceinline__ void edge_sd(const int* __restrict__ ei, int e, int E,
                                        int& src, int& dst) {
  if (e < E) { src = ei[e]; dst = ei[E + e]; }
  else       { src = e - E; dst = e - E; }
}

__global__ void k_edge_logit(const int* __restrict__ ei, const float* __restrict__ ea,
                             const float* __restrict__ loop_attr, const float* __restrict__ We,
                             const float* __restrict__ att, const float* __restrict__ xl,
                             const float* __restrict__ xr, float* __restrict__ logit,
                             unsigned* __restrict__ mb, int E, int N) {
  int t = blockIdx.x * blockDim.x + threadIdx.x;
  if (t >= (E + N) * HDS) return;
  int h = t & 3, e = t >> 2;
  int src, dst;
  edge_sd(ei, e, E, src, dst);
  const float* eav = (e < E) ? (ea + (size_t)e * 4) : (loop_attr + (size_t)src * 4);
  float ea0 = eav[0], ea1 = eav[1], ea2 = eav[2], ea3 = eav[3];
  const float* xls = xl + (size_t)src * DM + h * HIDC;
  const float* xrd = xr + (size_t)dst * DM + h * HIDC;
  __builtin_prefetch(xls, 0, 0);
  __builtin_prefetch(xrd, 0, 0);
  float acc = 0.f;
#pragma unroll 8
  for (int c = 0; c < HIDC; ++c) {
    int col = h * HIDC + c;
    float v = xls[c] + xrd[c] + ea0 * We[col] + ea1 * We[DM + col] + ea2 * We[2 * DM + col] +
              ea3 * We[3 * DM + col];
    v = (v > 0.f) ? v : NEGS * v;
    acc += v * att[col];
  }
  logit[(size_t)e * HDS + h] = acc;
  atomicMax(&mb[(size_t)dst * HDS + h], f2ord(acc));
}

__global__ void k_edge_expsum(const int* __restrict__ ei, float* __restrict__ p,
                              const unsigned* __restrict__ mb, float* __restrict__ s,
                              int E, int N) {
  int t = blockIdx.x * blockDim.x + threadIdx.x;
  if (t >= (E + N) * HDS) return;
  int h = t & 3, e = t >> 2;
  int src, dst;
  edge_sd(ei, e, E, src, dst);
  float lg = p[(size_t)e * HDS + h];
  float mx = ord2f(mb[(size_t)dst * HDS + h]);
  float pv = __expf(lg - mx);
  p[(size_t)e * HDS + h] = pv;
  atomicAdd(&s[(size_t)dst * HDS + h], pv);
}

__global__ void k_edge_scatter(const int* __restrict__ ei, const float* __restrict__ p,
                               const float* __restrict__ s, const float* __restrict__ xl,
                               float* __restrict__ ho, int E, int N) {
  int t = blockIdx.x * blockDim.x + threadIdx.x;
  if (t >= (E + N) * HDS) return;
  int h = t & 3, e = t >> 2;
  int src, dst;
  edge_sd(ei, e, E, src, dst);
  float alpha = p[(size_t)e * HDS + h] / (s[(size_t)dst * HDS + h] + 1e-16f);
  const float* xls = xl + (size_t)src * DM + h * HIDC;
  float* hod = ho + (size_t)dst * DM + h * HIDC;
  __builtin_prefetch(xls, 0, 0);
#pragma unroll 8
  for (int c = 0; c < HIDC; ++c) atomicAdd(&hod[c], alpha * xls[c]);
}

// mode 1: add bias, relu, emit bf16 for next layer.  mode 0: add bias in place (last layer).
__global__ void k_bias_act(float* __restrict__ hio, const float* __restrict__ bias,
                           bf16* __restrict__ xbf, int total, int relu) {
  int i = blockIdx.x * blockDim.x + threadIdx.x;
  if (i >= total) return;
  float v = hio[i] + bias[i & (DM - 1)];
  if (relu) {
    v = fmaxf(v, 0.f);
    xbf[i] = (bf16)v;
  } else {
    hio[i] = v;
  }
}

// ---------- pooling + final linear ----------
__global__ void k_pool_cnt(const int* __restrict__ batch, float* __restrict__ cntg, int N) {
  int i = blockIdx.x * blockDim.x + threadIdx.x;
  if (i < N) atomicAdd(&cntg[batch[i]], 1.0f);
}

__global__ void k_pool_accum(const int* __restrict__ batch, const float* __restrict__ ho,
                             float* __restrict__ pooled, int total) {
  int i = blockIdx.x * blockDim.x + threadIdx.x;
  if (i >= total) return;
  int node = i >> 7;
  atomicAdd(&pooled[(size_t)batch[node] * DM + (i & (DM - 1))], ho[i]);
}

__global__ void k_final(const float* __restrict__ pooled, const float* __restrict__ cntg,
                        const float* __restrict__ Wlin, const float* __restrict__ blin,
                        float* __restrict__ out, int G) {
  int g = blockIdx.x * blockDim.x + threadIdx.x;
  if (g >= G) return;
  float inv = 1.0f / fmaxf(cntg[g], 1.0f);
  float acc = blin[0];
#pragma unroll 8
  for (int c = 0; c < DM; ++c) acc += pooled[(size_t)g * DM + c] * inv * Wlin[c];
  out[g] = acc;
}

// ---------- host ----------
static inline unsigned cdiv(long a, long b) { return (unsigned)((a + b - 1) / b); }

extern "C" void kernel_launch(void* const* d_in, const int* in_sizes, int n_in,
                              void* d_out, int out_size, void* d_ws, size_t ws_size,
                              hipStream_t stream) {
  const float* x   = (const float*)d_in[0];
  const int* ei    = (const int*)d_in[1];
  const float* ea  = (const float*)d_in[2];
  const int* batch = (const int*)d_in[3];
  const int N = in_sizes[3];
  const int E = in_sizes[1] / 2;
  const int FIN = in_sizes[0] / N;
  const int ET = E + N;
  const int G = out_size;

  const float *Wl[3], *bl[3], *Wr[3], *br[3], *We[3], *att[3], *bias[3];
  for (int l = 0; l < 3; ++l) {
    int base = 4 + l * 7;
    Wl[l]   = (const float*)d_in[base + 0];
    bl[l]   = (const float*)d_in[base + 1];
    Wr[l]   = (const float*)d_in[base + 2];
    br[l]   = (const float*)d_in[base + 3];
    We[l]   = (const float*)d_in[base + 4];
    att[l]  = (const float*)d_in[base + 5];
    bias[l] = (const float*)d_in[base + 6];
  }
  const float* Wlin = (const float*)d_in[25];
  const float* blin = (const float*)d_in[26];
  float* out = (float*)d_out;

  // workspace carving (256B aligned)
  char* w = (char*)d_ws;
  auto carve = [&](size_t bytes) -> void* {
    void* p = (void*)w;
    w += (bytes + 255) & ~(size_t)255;
    return p;
  };
  float*    loop_attr = (float*)carve((size_t)N * 4 * 4);
  float*    cnt       = (float*)carve((size_t)N * 4);
  bf16*     xbf       = (bf16*)carve((size_t)N * DM * 2);
  bf16*     wlbf      = (bf16*)carve((size_t)DM * DM * 2);
  bf16*     wrbf      = (bf16*)carve((size_t)DM * DM * 2);
  float*    xl        = (float*)carve((size_t)N * DM * 4);
  float*    xr        = (float*)carve((size_t)N * DM * 4);
  unsigned* mb        = (unsigned*)carve((size_t)N * HDS * 4);
  float*    sb        = (float*)carve((size_t)N * HDS * 4);
  float*    pb        = (float*)carve((size_t)ET * HDS * 4);
  float*    ho        = (float*)carve((size_t)N * DM * 4);
  float*    pooled    = (float*)carve((size_t)G * DM * 4);
  float*    cntg      = (float*)carve((size_t)G * 4);

  const int B = 256;

  // self-loop edge attr = mean of incoming edge attrs
  hipMemsetAsync(loop_attr, 0, (size_t)N * 4 * 4, stream);
  hipMemsetAsync(cnt, 0, (size_t)N * 4, stream);
  k_loop_accum<<<cdiv(E, B), B, 0, stream>>>(ei, ea, loop_attr, cnt, E);
  k_loop_final<<<cdiv((long)N * 4, B), B, 0, stream>>>(loop_attr, cnt, N);

  // first-layer input in bf16
  k_cvt_bf16<<<cdiv((long)N * FIN, B), B, 0, stream>>>(x, xbf, N * FIN);

  int K = FIN;
  for (int l = 0; l < 3; ++l) {
    k_pack_w<<<cdiv((long)K * DM, B), B, 0, stream>>>(Wl[l], wlbf, K);
    k_pack_w<<<cdiv((long)K * DM, B), B, 0, stream>>>(Wr[l], wrbf, K);

    if (K == 64) {
      k_gemm_wmma<64><<<cdiv(N, 16), 32, 0, stream>>>(xbf, wlbf, wrbf, bl[l], br[l], xl, xr, N);
    } else {
      k_gemm_wmma<128><<<cdiv(N, 16), 32, 0, stream>>>(xbf, wlbf, wrbf, bl[l], br[l], xl, xr, N);
    }

    hipMemsetAsync(mb, 0, (size_t)N * HDS * 4, stream);
    hipMemsetAsync(sb, 0, (size_t)N * HDS * 4, stream);
    hipMemsetAsync(ho, 0, (size_t)N * DM * 4, stream);

    const long tot = (long)ET * HDS;
    k_edge_logit<<<cdiv(tot, B), B, 0, stream>>>(ei, ea, loop_attr, We[l], att[l], xl, xr,
                                                 pb, mb, E, N);
    k_edge_expsum<<<cdiv(tot, B), B, 0, stream>>>(ei, pb, mb, sb, E, N);
    k_edge_scatter<<<cdiv(tot, B), B, 0, stream>>>(ei, pb, sb, xl, ho, E, N);

    k_bias_act<<<cdiv((long)N * DM, B), B, 0, stream>>>(ho, bias[l], xbf, N * DM,
                                                        (l < 2) ? 1 : 0);
    K = DM;
  }

  // global mean pool + linear head
  hipMemsetAsync(pooled, 0, (size_t)G * DM * 4, stream);
  hipMemsetAsync(cntg, 0, (size_t)G * 4, stream);
  k_pool_cnt<<<cdiv(N, B), B, 0, stream>>>(batch, cntg, N);
  k_pool_accum<<<cdiv((long)N * DM, B), B, 0, stream>>>(batch, ho, pooled, N * DM);
  k_final<<<cdiv(G, B), B, 0, stream>>>(pooled, cntg, Wlin, blin, out, G);
}
